// GLCMLayer_32950989094982
// MI455X (gfx1250) — compile-verified, matrix operand order checked by reading, not run on hardware
//
#include <hip/hip_runtime.h>
#include <hip/hip_bf16.h>
#include <math.h>

// ---------------- problem constants ----------------
#define BATCH   4
#define TLEN    4096
#define DMODEL  1024
#define NPROJ   2048              // 2*D
#define MROWS   (BATCH*TLEN)      // 16384
#define KDIM    DMODEL            // 1024
#define EPSV    1e-8f

// GEMM tiling
#define BM 128
#define BN 64
#define BK 32
#define KITER (KDIM / BK)         // 32

typedef __attribute__((ext_vector_type(16))) __bf16        v16bf;
typedef __attribute__((ext_vector_type(8)))  float         v8f;
typedef __attribute__((ext_vector_type(4)))  unsigned int  u32x4;
typedef __attribute__((ext_vector_type(8)))  int           i32x8;
typedef __attribute__((ext_vector_type(4)))  int           i32x4;

#if defined(__has_builtin)
#  if __has_builtin(__builtin_amdgcn_tensor_load_to_lds)
#    define USE_TDM 1
#  endif
#endif
#ifndef USE_TDM
#  define USE_TDM 0
#endif

// round-to-nearest-even f32 -> bf16 raw bits
__device__ __forceinline__ unsigned short f2bf(float f) {
    union { float f; unsigned u; } v; v.f = f;
    unsigned r = v.u + 0x7FFFu + ((v.u >> 16) & 1u);
    return (unsigned short)(r >> 16);
}

__device__ __forceinline__ float sigm(float x) {
    return 1.0f / (1.0f + __expf(-x));
}

#if USE_TDM
// Issue a TDM 2-D tile load: tile_d1 rows x tile_d0 elements (2-byte elems),
// row stride d0_stride elements, packed contiguously into LDS at lds_off.
// Descriptor bitfields per CDNA5 ISA 8.3/8.4 (groups 2/3 zero: tile_dim2=0 unused).
__device__ __forceinline__ void tdm_load_2d(const void* gaddr, unsigned lds_off,
                                            unsigned tensor_d0, unsigned tensor_d1,
                                            unsigned tile_d0, unsigned tile_d1,
                                            unsigned long long d0_stride) {
    unsigned long long ga = (unsigned long long)(size_t)gaddr;
    u32x4 g0;
    g0[0] = 1u;                                              // count=1 (valid), user mode
    g0[1] = lds_off;                                         // lds_addr
    g0[2] = (unsigned)(ga & 0xFFFFFFFFu);                    // global_addr[31:0]
    g0[3] = (unsigned)((ga >> 32) & 0x01FFFFFFu) | (2u << 30); // global_addr[56:32] | type=2
    i32x8 g1;
    g1[0] = (int)(1u << 16);                                 // data_size=1 (2 bytes)
    g1[1] = (int)((tensor_d0 & 0xFFFFu) << 16);              // tensor_dim0[15:0]
    g1[2] = (int)(((tensor_d0 >> 16) & 0xFFFFu) |
                  ((tensor_d1 & 0xFFFFu) << 16));            // tensor_dim0[31:16] | tensor_dim1[15:0]
    g1[3] = (int)(((tensor_d1 >> 16) & 0xFFFFu) |
                  ((tile_d0 & 0xFFFFu) << 16));              // tensor_dim1[31:16] | tile_dim0
    g1[4] = (int)(tile_d1 & 0xFFFFu);                        // tile_dim1 (tile_dim2=0)
    g1[5] = (int)(unsigned)(d0_stride & 0xFFFFFFFFull);      // tensor_dim0_stride[31:0]
    g1[6] = (int)(unsigned)((d0_stride >> 32) & 0xFFFFull);  // stride[47:32] (dim1_stride=0)
    g1[7] = 0;
    i32x4 z4 = {0, 0, 0, 0};
#if __clang_major__ >= 23
    i32x8 z8 = {0, 0, 0, 0, 0, 0, 0, 0};
    __builtin_amdgcn_tensor_load_to_lds(g0, g1, z4, z4, z8, 0);
#else
    __builtin_amdgcn_tensor_load_to_lds(g0, g1, z4, z4, 0);
#endif
}
#endif // USE_TDM

// ---------------- stage 1: x (f32) -> bf16 ----------------
__global__ __launch_bounds__(256) void cvt_x_bf16(const float* __restrict__ x,
                                                  unsigned short* __restrict__ xb) {
    size_t i = ((size_t)blockIdx.x * 256 + threadIdx.x) * 4;
    if (i + 3 >= (size_t)MROWS * KDIM) return;
    float4 v = *(const float4*)(x + i);
    ushort4 o;
    o.x = f2bf(v.x); o.y = f2bf(v.y); o.z = f2bf(v.z); o.w = f2bf(v.w);
    *(ushort4*)(xb + i) = o;
}

// ---------------- stage 2: W_proj (D x 2D, f32) -> W^T (2D x D, bf16) ----------------
__global__ __launch_bounds__(256) void transpose_w_bf16(const float* __restrict__ W,
                                                        unsigned short* __restrict__ WT) {
    __shared__ float tile[32][33];
    int fx = blockIdx.x * 32 + threadIdx.x;
    int dy = blockIdx.y * 32;
    #pragma unroll
    for (int j = 0; j < 4; ++j) {
        int d = dy + threadIdx.y + j * 8;
        tile[threadIdx.y + j * 8][threadIdx.x] = W[(size_t)d * NPROJ + fx];
    }
    __syncthreads();
    int dout = dy + threadIdx.x;
    #pragma unroll
    for (int j = 0; j < 4; ++j) {
        int fout = blockIdx.x * 32 + threadIdx.y + j * 8;
        WT[(size_t)fout * KDIM + dout] = f2bf(tile[threadIdx.x][threadIdx.y + j * 8]);
    }
}

// ---------------- stage 3: bf16 WMMA GEMM  proj = x @ W + b ----------------
// A: MROWS x KDIM row-major bf16.  B: NPROJ x KDIM row-major (= W^T) bf16.
// grid (NPROJ/BN, MROWS/BM); 256 threads = 8 waves; wave w owns M rows [w*16, w*16+16).
// Data path: TDM double-buffered tile loads (wave0 issues, s_wait_tensorcnt + barrier
// publishes), WMMA consumes fragments straight from LDS.
__global__ __launch_bounds__(256) void gemm_wmma_bf16(const unsigned short* __restrict__ A,
                                                      const unsigned short* __restrict__ BT,
                                                      const float* __restrict__ bias,
                                                      float* __restrict__ C) {
    const int tid   = threadIdx.x;
    const int wave  = tid >> 5;
    const int lane  = tid & 31;
    const int m16   = lane & 15;
    const int hf    = lane >> 4;
    const int waveM = wave * 16;

    const int tileM = blockIdx.y * BM;
    const int tileN = blockIdx.x * BN;

    v8f acc[4];
    #pragma unroll
    for (int i = 0; i < 4; ++i) acc[i] = (v8f){0.f,0.f,0.f,0.f,0.f,0.f,0.f,0.f};

#if USE_TDM
    __shared__ __align__(16) unsigned short As[2][BM][BK];   // 16 KB
    __shared__ __align__(16) unsigned short Bs[2][BN][BK];   //  8 KB

    const unsigned short* Abase = A  + (size_t)tileM * KDIM;
    const unsigned short* Bbase = BT + (size_t)tileN * KDIM;
    const unsigned asoff0 = (unsigned)(size_t)&As[0][0][0];
    const unsigned asoff1 = (unsigned)(size_t)&As[1][0][0];
    const unsigned bsoff0 = (unsigned)(size_t)&Bs[0][0][0];
    const unsigned bsoff1 = (unsigned)(size_t)&Bs[1][0][0];

    if (tid < 32) {   // wave 0: prologue TDM for tile 0
        tdm_load_2d(Abase, asoff0, KDIM, MROWS, BK, BM, KDIM);
        tdm_load_2d(Bbase, bsoff0, KDIM, NPROJ, BK, BN, KDIM);
    }

    for (int it = 0; it < KITER; ++it) {
        const int cur = it & 1;
        __builtin_amdgcn_s_wait_tensorcnt(0);   // wave0: tile `it` landed; others no-op
        __syncthreads();                        // publish LDS; all prior reads of other buf done
        if (it + 1 < KITER && tid < 32) {       // kick tile it+1 into the other buffer
            const int kb = (it + 1) * BK;
            const unsigned ao = (it + 1) & 1 ? asoff1 : asoff0;
            const unsigned bo = (it + 1) & 1 ? bsoff1 : bsoff0;
            tdm_load_2d(Abase + kb, ao, KDIM, MROWS, BK, BM, KDIM);
            tdm_load_2d(Bbase + kb, bo, KDIM, NPROJ, BK, BN, KDIM);
        }

        // A fragment: lanes 0-15 -> K[0..7]+K[16..23], lanes 16-31 -> K[8..15]+K[24..31]
        v16bf afrag;
        *(uint4*)&afrag       = *(const uint4*)&As[cur][waveM + m16][hf * 8];
        *((uint4*)&afrag + 1) = *(const uint4*)&As[cur][waveM + m16][16 + hf * 8];
        #pragma unroll
        for (int nt = 0; nt < 4; ++nt) {
            v16bf bfrag;
            *(uint4*)&bfrag       = *(const uint4*)&Bs[cur][nt * 16 + m16][hf * 8];
            *((uint4*)&bfrag + 1) = *(const uint4*)&Bs[cur][nt * 16 + m16][16 + hf * 8];
            acc[nt] = __builtin_amdgcn_wmma_f32_16x16x32_bf16(
                false, afrag, false, bfrag, (short)0, acc[nt], false, false);
        }
    }
#else
    // fallback: cooperative global->LDS staging (round-1 path)
    __shared__ __align__(16) unsigned short As[BM][BK];
    __shared__ __align__(16) unsigned short Bs[BN][BK];
    const int ar = tid >> 1, ac = (tid & 1) * 16;
    const int br = tid >> 2, bc = (tid & 3) * 8;
    for (int kb = 0; kb < KDIM; kb += BK) {
        *(uint4*)&As[ar][ac] = *(const uint4*)&A[(size_t)(tileM + ar) * KDIM + kb + ac];
        *(uint4*)&Bs[br][bc] = *(const uint4*)&BT[(size_t)(tileN + br) * KDIM + kb + bc];
        if (kb + BK < KDIM) {
            __builtin_prefetch(&A[(size_t)(tileM + ar) * KDIM + kb + BK + ac], 0, 0);
            __builtin_prefetch(&BT[(size_t)(tileN + br) * KDIM + kb + BK + bc], 0, 0);
        }
        __syncthreads();
        v16bf afrag;
        *(uint4*)&afrag       = *(const uint4*)&As[waveM + m16][hf * 8];
        *((uint4*)&afrag + 1) = *(const uint4*)&As[waveM + m16][16 + hf * 8];
        #pragma unroll
        for (int nt = 0; nt < 4; ++nt) {
            v16bf bfrag;
            *(uint4*)&bfrag       = *(const uint4*)&Bs[nt * 16 + m16][hf * 8];
            *((uint4*)&bfrag + 1) = *(const uint4*)&Bs[nt * 16 + m16][16 + hf * 8];
            acc[nt] = __builtin_amdgcn_wmma_f32_16x16x32_bf16(
                false, afrag, false, bfrag, (short)0, acc[nt], false, false);
        }
        __syncthreads();
    }
#endif

    // epilogue: add bias, store f32 tile.
    // C layout: VGPR r -> row (r + 8*laneHalf), col = lane&15 within each 16x16 tile
    #pragma unroll
    for (int nt = 0; nt < 4; ++nt) {
        const int n = tileN + nt * 16 + m16;
        const float bv = bias[n];
        #pragma unroll
        for (int r = 0; r < 8; ++r) {
            const int mrow = tileM + waveM + r + hf * 8;
            C[(size_t)mrow * NPROJ + n] = acc[nt][r] + bv;
        }
    }
}

// ---------------- stage 4: in-place RoPE on proj[:, :D], sigmoid on proj[:, D:] ----------
__global__ __launch_bounds__(256) void rope_gate(float* __restrict__ proj,
                                                 const float* __restrict__ sinT,
                                                 const float* __restrict__ cosT) {
    size_t idx = (size_t)blockIdx.x * 256 + threadIdx.x;
    if (idx >= (size_t)MROWS * (DMODEL / 2)) return;
    const int row = (int)(idx >> 9);
    const int i   = (int)(idx & 511);
    const int t   = row & (TLEN - 1);

    float* p = proj + (size_t)row * NPROJ;
    const float x1 = p[2 * i];
    const float x2 = p[2 * i + 1];
    const float c  = cosT[(size_t)t * (DMODEL / 2) + i];
    const float s  = sinT[(size_t)t * (DMODEL / 2) + i];
    p[2 * i]     = x1 * c - x2 * s;
    p[2 * i + 1] = x1 * s + x2 * c;

    float g1 = p[DMODEL + 2 * i];
    float g2 = p[DMODEL + 2 * i + 1];
    p[DMODEL + 2 * i]     = sigm(g1);
    p[DMODEL + 2 * i + 1] = sigm(g2);
}

// ---------------- stage 5: depthwise conv7 + SiLU + gate + RMSNorm -------------------
__global__ __launch_bounds__(256) void tail_fused(const float* __restrict__ proj,
                                                  const float* __restrict__ dwk,
                                                  const float* __restrict__ dwb,
                                                  const float* __restrict__ gamma,
                                                  float* __restrict__ out) {
    const int row = blockIdx.x;
    const int b   = row >> 12;
    const int t   = row & (TLEN - 1);
    const int d   = threadIdx.x * 4;

    const float* base = proj + (size_t)b * TLEN * NPROJ;
    float a0 = 0.f, a1 = 0.f, a2 = 0.f, a3 = 0.f;
    #pragma unroll
    for (int k = 0; k < 7; ++k) {
        const int tt = t + k - 3;
        if (tt < 0 || tt >= TLEN) continue;
        const float* xr = base + (size_t)tt * NPROJ + d;
        const float* w  = dwk + (size_t)k * DMODEL + d;
        a0 += xr[0] * w[0]; a1 += xr[1] * w[1];
        a2 += xr[2] * w[2]; a3 += xr[3] * w[3];
    }
    a0 += dwb[d + 0]; a1 += dwb[d + 1]; a2 += dwb[d + 2]; a3 += dwb[d + 3];

    a0 *= sigm(a0); a1 *= sigm(a1); a2 *= sigm(a2); a3 *= sigm(a3);

    const float* gp = proj + (size_t)row * NPROJ + DMODEL + d;
    const float g0 = a0 * gp[0];
    const float g1 = a1 * gp[1];
    const float g2 = a2 * gp[2];
    const float g3 = a3 * gp[3];

    __shared__ float red[256];
    red[threadIdx.x] = g0 * g0 + g1 * g1 + g2 * g2 + g3 * g3;
    __syncthreads();
    #pragma unroll
    for (int s = 128; s > 0; s >>= 1) {
        if (threadIdx.x < s) red[threadIdx.x] += red[threadIdx.x + s];
        __syncthreads();
    }
    const float inv = rsqrtf(red[0] * (1.0f / (float)DMODEL) + EPSV);

    float* o = out + (size_t)row * DMODEL + d;
    o[0] = gamma[d + 0] * g0 * inv;
    o[1] = gamma[d + 1] * g1 * inv;
    o[2] = gamma[d + 2] * g2 * inv;
    o[3] = gamma[d + 3] * g3 * inv;
}

// ---------------- launcher ----------------
extern "C" void kernel_launch(void* const* d_in, const int* in_sizes, int n_in,
                              void* d_out, int out_size, void* d_ws, size_t ws_size,
                              hipStream_t stream) {
    const float* x    = (const float*)d_in[0];
    const float* sinT = (const float*)d_in[1];
    const float* cosT = (const float*)d_in[2];
    const float* W    = (const float*)d_in[3];
    const float* bpr  = (const float*)d_in[4];
    const float* dwk  = (const float*)d_in[5];
    const float* dwb  = (const float*)d_in[6];
    const float* gam  = (const float*)d_in[7];
    float* out = (float*)d_out;

    unsigned short* xbf  = (unsigned short*)d_ws;                     // 32 MB
    unsigned short* wtbf = xbf + (size_t)MROWS * KDIM;                //  4 MB
    float* proj = (float*)((char*)d_ws + (size_t)36 * 1024 * 1024);   // 128 MB

    {
        size_t n4 = (size_t)MROWS * KDIM / 4;
        cvt_x_bf16<<<(unsigned)((n4 + 255) / 256), 256, 0, stream>>>(x, xbf);
    }
    {
        dim3 g(NPROJ / 32, KDIM / 32), blk(32, 8);
        transpose_w_bf16<<<g, blk, 0, stream>>>(W, wtbf);
    }
    {
        dim3 g(NPROJ / BN, MROWS / BM);
        gemm_wmma_bf16<<<g, 256, 0, stream>>>(xbf, wtbf, bpr, proj);
    }
    {
        size_t n = (size_t)MROWS * (DMODEL / 2);
        rope_gate<<<(unsigned)((n + 255) / 256), 256, 0, stream>>>(proj, sinT, cosT);
    }
    tail_fused<<<MROWS, 256, 0, stream>>>(proj, dwk, dwb, gam, out);
}
